// LSTM_36069135352174
// MI455X (gfx1250) — compile-verified, matrix operand order checked by reading, not run on hardware
//
#include <hip/hip_runtime.h>
#include <math.h>

// LSTM: B=64, T=512, I=H=1024. Mixed precision: bf16 WMMA operands,
// fp32 accumulate / cell state / gate math (v_wmma_f32_16x16x32_bf16).
// Recurrence uses the Tensor Data Mover to stage h_prev into LDS once per
// block per step (TENSORcnt), A fragments then come from LDS (ds_load_b128).
#define B_  64
#define T_  512
#define I_  1024
#define H_  1024
#define G4_ 4096   // 4*H

typedef __bf16 bf16_t;
typedef __attribute__((ext_vector_type(16))) __bf16 v16bf;
typedef __attribute__((ext_vector_type(8)))  float  v8f;
typedef __attribute__((ext_vector_type(4)))  unsigned v4u;
typedef __attribute__((ext_vector_type(8)))  int    v8i;
typedef __attribute__((ext_vector_type(4)))  int    v4i;

union FragU {
  v16bf v;
  uint4 u[2];
};

// Load one 16x32 bf16 WMMA operand fragment from global memory.
// Layout (CDNA5 ISA 7.12.2, 16-bit A 16x32): lane<16 -> row=lane, K-base 0;
// lane>=16 -> row=lane-16, K-base 8. VGPRs 0..3 hold K=kb..kb+7 (contiguous
// halves), VGPRs 4..7 hold K=16+kb..16+kb+7. Each lane loads two contiguous
// 16-byte runs -> two global_load_b128.
// Works for A (row-major [M,K]) and for B when B is stored N-major ([N,K]).
static __device__ __forceinline__ v16bf load_frag16x32(const bf16_t* p, int ld) {
  const int lane = threadIdx.x & 31;
  const int r    = lane & 15;
  const int kb   = (lane >> 4) << 3;   // 0 or 8
  const bf16_t* q = p + (long)r * ld + kb;
  FragU f;
  f.u[0] = *(const uint4*)(q);         // K = kb .. kb+7
  f.u[1] = *(const uint4*)(q + 16);    // K = 16+kb .. 16+kb+7
  return f.v;
}

static __device__ __forceinline__ v8f wmma_bf16(v16bf a, v16bf b, v8f c) {
  // 8 args: (neg_a, A, neg_b, B, c_mod, C, reuse_a, reuse_b)
  return __builtin_amdgcn_wmma_f32_16x16x32_bf16(false, a, false, b,
                                                 (short)0, c, false, false);
}

static __device__ __forceinline__ float sigmoidf_(float x) {
  return 1.0f / (1.0f + __expf(-x));
}

// ---- pack x: fp32 [B,T,I] -> bf16 [T-1, B, I] (timestep-major) ------------
__global__ void lstm_pack_x(const float* __restrict__ x, bf16_t* __restrict__ xbf) {
  long idx = (long)blockIdx.x * blockDim.x + threadIdx.x;
  const long n4 = (long)(T_ - 1) * B_ * I_ / 4;
  if (idx >= n4) return;
  long o   = idx * 4;
  int  t   = (int)(o / ((long)B_ * I_));
  int  rem = (int)(o % ((long)B_ * I_));
  int  b   = rem / I_;
  int  i   = rem % I_;
  const float4 v = *(const float4*)(x + ((long)b * T_ + t) * I_ + i);
  union { bf16_t h[4]; uint2 u; } o4;
  o4.h[0] = (bf16_t)v.x; o4.h[1] = (bf16_t)v.y;
  o4.h[2] = (bf16_t)v.z; o4.h[3] = (bf16_t)v.w;
  *(uint2*)(xbf + o) = o4.u;
}

// ---- pack weights: Ut[4H, I] = U^T (bf16), Vt[4H, H] = V^T (bf16) ---------
__global__ void lstm_pack_w(const float* __restrict__ U_i, const float* __restrict__ U_f,
                            const float* __restrict__ U_c, const float* __restrict__ U_o,
                            const float* __restrict__ V_i, const float* __restrict__ V_f,
                            const float* __restrict__ V_c, const float* __restrict__ V_o,
                            bf16_t* __restrict__ Ut, bf16_t* __restrict__ Vt) {
  long idx = (long)blockIdx.x * blockDim.x + threadIdx.x;
  const long NW = (long)G4_ * H_;           // elements per transposed matrix
  if (idx >= 2 * NW) return;
  bool isV = (idx >= NW);
  long l = isV ? (idx - NW) : idx;
  int  g = (int)(l / H_);                   // output row in [0, 4H)
  int  k = (int)(l % H_);                   // output col (K index)
  int  q = g >> 10;                         // gate
  int  n = g & 1023;                        // column within gate
  const float* src;
  if (isV) src = (q == 0) ? V_i : (q == 1) ? V_f : (q == 2) ? V_c : V_o;
  else     src = (q == 0) ? U_i : (q == 1) ? U_f : (q == 2) ? U_c : U_o;
  bf16_t* dst = isV ? Vt : Ut;
  dst[l] = (bf16_t)src[(long)k * H_ + n];   // transpose: [K,N] -> [N,K]
}

// ---- pack bias: concat 4x [H] -> [4H] fp32 --------------------------------
__global__ void lstm_pack_b(const float* __restrict__ b_i, const float* __restrict__ b_f,
                            const float* __restrict__ b_c, const float* __restrict__ b_o,
                            float* __restrict__ bias) {
  int g = blockIdx.x * blockDim.x + threadIdx.x;
  if (g >= G4_) return;
  int q = g >> 10, n = g & 1023;
  const float* s = (q == 0) ? b_i : (q == 1) ? b_f : (q == 2) ? b_c : b_o;
  bias[g] = s[n];
}

// ---- zero state (h ping-pong bf16, c fp32) --------------------------------
__global__ void lstm_init_state(float* __restrict__ c,
                                bf16_t* __restrict__ h0, bf16_t* __restrict__ h1) {
  int i = blockIdx.x * blockDim.x + threadIdx.x;
  if (i < B_ * H_) {
    c[i]  = 0.0f;
    h0[i] = (bf16_t)0.0f;
    h1[i] = (bf16_t)0.0f;
  }
}

// ---- x_proj = x @ U + b : WMMA GEMM, M=(T-1)*B, K=I, N=4H -----------------
// 128 threads = 4 waves; block tile 64x64; wave tile 32x32 (2x2 WMMA tiles).
__global__ void lstm_gemm_xU(const bf16_t* __restrict__ A,   // [M, K] row-major
                             const bf16_t* __restrict__ Bt,  // [N, K] (U^T)
                             const float*  __restrict__ bias,// [N]
                             float* __restrict__ C,          // [M, N]
                             int M, int N, int K) {
  const int bm   = blockIdx.x * 64;
  const int bn   = blockIdx.y * 64;
  const int wave = threadIdx.x >> 5;
  const int wr   = (wave >> 1) * 32;
  const int wc   = (wave & 1) * 32;
  const int lane = threadIdx.x & 31;

  v8f acc[2][2] = {};
  const bf16_t* Ab = A  + (long)(bm + wr) * K;
  const bf16_t* Bb = Bt + (long)(bn + wc) * K;

  for (int k = 0; k < K; k += 32) {
    __builtin_prefetch(Ab + k + 256, 0, 0);           // global_prefetch_b8
    v16bf a0 = load_frag16x32(Ab + k,            K);
    v16bf a1 = load_frag16x32(Ab + (long)16 * K + k, K);
    v16bf b0 = load_frag16x32(Bb + k,            K);
    v16bf b1 = load_frag16x32(Bb + (long)16 * K + k, K);
    acc[0][0] = wmma_bf16(a0, b0, acc[0][0]);
    acc[0][1] = wmma_bf16(a0, b1, acc[0][1]);
    acc[1][0] = wmma_bf16(a1, b0, acc[1][0]);
    acc[1][1] = wmma_bf16(a1, b1, acc[1][1]);
  }

  // C/D layout: VGPR v -> M = v + 8*(lane>=16); N = lane&15.
  const int mo = (lane >> 4) << 3;
  const int nc = lane & 15;
#pragma unroll
  for (int i = 0; i < 2; ++i) {
#pragma unroll
    for (int j = 0; j < 2; ++j) {
      const int gn = bn + wc + j * 16 + nc;
      const float bv = bias[gn];
      const int gm0 = bm + wr + i * 16 + mo;
#pragma unroll
      for (int v = 0; v < 8; ++v)
        C[(long)(gm0 + v) * N + gn] = acc[i][j][v] + bv;
    }
  }
}

// ---- one recurrence step: z = x_proj[t] + h@V; gates; c,h update ----------
// Grid: H/16 = 64 blocks. 128 threads = 4 waves; wave q = gate q.
// h_prev (64x1024 bf16 = 128 KB) is DMA'd into LDS once per block by the
// Tensor Data Mover (TENSORcnt); A fragments then read via ds_load_b128,
// eliminating the 4x-redundant global A traffic on the serial critical path.
// Each wave: 64x16 strip of z for its gate (4 WMMA row tiles, K-loop 1024).
// Gates exchanged via LDS, then pointwise LSTM math on 64x16 elements.
__global__ void lstm_step(const float*  __restrict__ xp,    // [B, 4H] slice at t
                          const bf16_t* __restrict__ Vt,    // [4H, H] (V^T)
                          const bf16_t* __restrict__ hprev, // [B, H] bf16
                          bf16_t* __restrict__ hnext,       // [B, H] bf16
                          float* __restrict__ c,            // [B, H] fp32
                          float* __restrict__ hout) {       // null or [B, H] fp32
  __shared__ bf16_t hsh[B_ * H_];      // 128 KB staged h_prev (TDM target)
  __shared__ float  zl[4][B_][16];     // 16 KB gate exchange
  const int q    = threadIdx.x >> 5;   // gate index 0..3
  const int lane = threadIdx.x & 31;
  const int jc   = blockIdx.x * 16;    // column group within H

  // --- TDM: DMA h_prev tile into LDS (wave 0 issues; others wait at barrier)
  if (threadIdx.x < 32) {
    const unsigned lds_base = (unsigned)(uintptr_t)(&hsh[0]); // LDS byte addr
    const unsigned long long ga = (unsigned long long)(uintptr_t)hprev;
    // D# group 0: count=1 | lds_addr | global_addr[56:0] | type=2
    v4u g0 = { 1u,
               lds_base,
               (unsigned)(ga & 0xFFFFFFFFu),
               (unsigned)((ga >> 32) & 0x01FFFFFFu) | 0x80000000u };
    // D# group 1: data_size=1(2B); tensor_dim0=1024; tensor_dim1=64;
    // tile_dim0=1024; tile_dim1=64; tensor_dim0_stride=1024.
    v8i g1 = { (int)0x00010000,        // [17:16] data_size = 2 bytes
               (int)(1024u << 16),     // tensor_dim0[15:0]  -> bits [63:48]
               (int)(64u   << 16),     // tensor_dim1[15:0]  -> bits [95:80], [31:16] of word
               (int)(1024u << 16),     // tile_dim0          -> bits [127:112]
               (int)64,                // tile_dim1          -> bits [143:128]
               (int)1024,              // tensor_dim0_stride low 32
               0, 0 };
    v4i z4 = { 0, 0, 0, 0 };
#if __clang_major__ >= 23
    v8i z8 = { 0, 0, 0, 0, 0, 0, 0, 0 };
    __builtin_amdgcn_tensor_load_to_lds(g0, g1, z4, z4, z8, 0);
#else
    __builtin_amdgcn_tensor_load_to_lds(g0, g1, z4, z4, 0);
#endif
    __builtin_amdgcn_s_wait_tensorcnt(0);
  }
  __syncthreads();

  v8f acc[4] = {};
  const bf16_t* Bb = Vt + ((long)q * H_ + jc) * H_;
  const int r  = lane & 15;
  const int kb = (lane >> 4) << 3;

  for (int k = 0; k < H_; k += 32) {
    v16bf b = load_frag16x32(Bb + k, H_);   // V fragment from global/L2
#pragma unroll
    for (int m = 0; m < 4; ++m) {           // A fragments from LDS (ds_load_b128)
      FragU f;
      const int base = (m * 16 + r) * H_ + k + kb;
      f.u[0] = *(const uint4*)&hsh[base];
      f.u[1] = *(const uint4*)&hsh[base + 16];
      acc[m] = wmma_bf16(f.v, b, acc[m]);
    }
  }

  const int mo = (lane >> 4) << 3;
  const int nc = lane & 15;
#pragma unroll
  for (int m = 0; m < 4; ++m) {
#pragma unroll
    for (int v = 0; v < 8; ++v) {
      const int row = m * 16 + mo + v;
      zl[q][row][nc] = acc[m][v] + xp[(long)row * G4_ + q * H_ + jc + nc];
    }
  }
  __syncthreads();

  for (int idx = threadIdx.x; idx < B_ * 16; idx += 128) {
    const int row = idx >> 4;
    const int col = idx & 15;
    const float ig = sigmoidf_(zl[0][row][col]);
    const float fg = sigmoidf_(zl[1][row][col]);
    const float gg = tanhf(zl[2][row][col]);
    const float og = sigmoidf_(zl[3][row][col]);
    const long  ci = (long)row * H_ + jc + col;
    const float cn = fg * c[ci] + ig * gg;
    c[ci] = cn;
    const float hn = og * tanhf(cn);
    hnext[ci] = (bf16_t)hn;
    if (hout) hout[ci] = hn;
  }
}

extern "C" void kernel_launch(void* const* d_in, const int* in_sizes, int n_in,
                              void* d_out, int out_size, void* d_ws, size_t ws_size,
                              hipStream_t stream) {
  (void)in_sizes; (void)n_in; (void)out_size; (void)ws_size;

  const float* x   = (const float*)d_in[0];
  const float* U_i = (const float*)d_in[1];
  const float* V_i = (const float*)d_in[2];
  const float* b_i = (const float*)d_in[3];
  const float* U_f = (const float*)d_in[4];
  const float* V_f = (const float*)d_in[5];
  const float* b_f = (const float*)d_in[6];
  const float* U_c = (const float*)d_in[7];
  const float* V_c = (const float*)d_in[8];
  const float* b_c = (const float*)d_in[9];
  const float* U_o = (const float*)d_in[10];
  const float* V_o = (const float*)d_in[11];
  const float* b_o = (const float*)d_in[12];
  float* out = (float*)d_out;

  // Workspace layout (~620 MB): xbf 67MB, xproj 536MB, Ut/Vt 8MB each,
  // bias, h ping-pong (bf16), c (fp32).
  char* base = (char*)d_ws;
  size_t off = 0;
  auto take = [&](size_t bytes) -> char* {
    char* p = base + off;
    off = (off + bytes + 255) & ~(size_t)255;
    return p;
  };
  bf16_t* xbf  = (bf16_t*)take((size_t)(T_ - 1) * B_ * I_ * sizeof(bf16_t));
  float*  xprj = (float*) take((size_t)(T_ - 1) * B_ * G4_ * sizeof(float));
  bf16_t* Ut   = (bf16_t*)take((size_t)G4_ * I_ * sizeof(bf16_t));
  bf16_t* Vt   = (bf16_t*)take((size_t)G4_ * H_ * sizeof(bf16_t));
  float*  bias = (float*) take((size_t)G4_ * sizeof(float));
  bf16_t* hbf0 = (bf16_t*)take((size_t)B_ * H_ * sizeof(bf16_t));
  bf16_t* hbf1 = (bf16_t*)take((size_t)B_ * H_ * sizeof(bf16_t));
  float*  cst  = (float*) take((size_t)B_ * H_ * sizeof(float));

  // 1) Pack / convert.
  {
    long n4 = (long)(T_ - 1) * B_ * I_ / 4;
    lstm_pack_x<<<(int)((n4 + 255) / 256), 256, 0, stream>>>(x, xbf);
    long nw = 2L * G4_ * H_;
    lstm_pack_w<<<(int)((nw + 255) / 256), 256, 0, stream>>>(
        U_i, U_f, U_c, U_o, V_i, V_f, V_c, V_o, Ut, Vt);
    lstm_pack_b<<<(G4_ + 255) / 256, 256, 0, stream>>>(b_i, b_f, b_c, b_o, bias);
    lstm_init_state<<<(B_ * H_ + 255) / 256, 256, 0, stream>>>(cst, hbf0, hbf1);
  }

  // 2) Input projection GEMM: x_proj = x@U + b. M=32704, N=4096, K=1024.
  {
    const int M = (T_ - 1) * B_, N = G4_, K = I_;
    dim3 grid(M / 64, N / 64);
    lstm_gemm_xU<<<grid, 128, 0, stream>>>(xbf, Ut, bias, xprj, M, N, K);
  }

  // 3) Recurrence: 511 scan steps + 1 repeat of the last timestep (quirk).
  //    h ping-pongs between bf16 buffers; stream ordering serializes steps.
  bf16_t* hb[2] = {hbf0, hbf1};
  for (int s = 0; s < T_; ++s) {
    const int t = (s < T_ - 1) ? s : (T_ - 2);
    const float* xp = xprj + (size_t)t * B_ * G4_;
    float* hout = (s == T_ - 1) ? out : nullptr;
    lstm_step<<<H_ / 16, 128, 0, stream>>>(xp, Vt, hb[s & 1], hb[(s + 1) & 1],
                                           cst, hout);
  }
}
